// SimplifiedLlama4Attention_87771951661436
// MI455X (gfx1250) — matrix operand order repaired and checked by reading, not compiled, not run on hardware
//
#include <hip/hip_runtime.h>
#include <hip/hip_bf16.h>
#include <type_traits>

#define DEVI __device__ __forceinline__

typedef __attribute__((ext_vector_type(16))) __bf16          v16bf;
typedef __attribute__((ext_vector_type(8)))  float           v8f;
typedef __attribute__((ext_vector_type(8)))  unsigned short  us8;
typedef __attribute__((ext_vector_type(16))) unsigned short  us16;
typedef __attribute__((ext_vector_type(4)))  float           f4;

static constexpr int NB  = 2;     // batch
static constexpr int NS  = 2048;  // seq len
static constexpr int ND  = 2048;  // model dim
static constexpr int NH  = 16;    // q heads
static constexpr int NKV = 4;     // kv heads
static constexpr int HD  = 128;   // head dim

DEVI unsigned short f2bf(float f) {
    __bf16 h = (__bf16)f;                       // RNE; lowers to v_cvt_*bf16_f32
    return __builtin_bit_cast(unsigned short, h);
}

DEVI v16bf mk16(us8 lo, us8 hi) {
    us16 w;
#pragma unroll
    for (int i = 0; i < 8; ++i) { w[i] = lo[i]; w[i + 8] = hi[i]; }
    return __builtin_bit_cast(v16bf, w);
}

// ---------------------------------------------------------------------------
// Generic batched GEMM: C(f32)[M,N] = alpha * A[M,K] @ B[K,N]  (+causal mask)
// A/B are f32 (converted to bf16 while staging to LDS) or raw bf16 bits.
// Batch z decomposes as z = zo*Hdiv + zi; per-operand offset =
//   zo*O + (zi/rep)*I   (rep implements GQA head sharing).
// Tiles: BM=64, BN=128, BK=64. 256 threads = 8 waves; wave = 16x64 of C.
// Software-pipelined: next K-tile is loaded to VGPRs while WMMAs run.
// ---------------------------------------------------------------------------
template <typename TA, typename TB>
__global__ __launch_bounds__(256)
void gemm_wmma(const TA* __restrict__ A, long lda, long aO, long aI, int aRep,
               const TB* __restrict__ Bm, long ldb, long bO, long bI, int bRep,
               float* __restrict__ C, long ldc, long cO, long cI,
               int K, int Hdiv, float alpha, int causal)
{
    constexpr int BM = 64, BN = 128, BK = 64, BKP = 72; // pad keeps 16B align
    __shared__ unsigned short As[BM][BKP];  // [m][k] bf16 bits
    __shared__ unsigned short Bs[BN][BKP];  // [n][k] bf16 bits (B transposed)

    const int z  = blockIdx.z;
    const int zo = z / Hdiv, zi = z % Hdiv;
    A  += (long)zo * aO + (long)(zi / aRep) * aI;
    Bm += (long)zo * bO + (long)(zi / bRep) * bI;
    C  += (long)zo * cO + (long)zi * cI;

    const int tid  = threadIdx.x;
    const int lane = tid & 31;
    const int wave = tid >> 5;
    const int g    = lane >> 4;      // half-wave select
    const int sub  = lane & 15;
    const int wm   = (wave >> 1) * 16;   // wave M offset in block tile
    const int wn   = (wave & 1) * 64;    // wave N offset in block tile
    const int m0   = blockIdx.y * BM;
    const int n0   = blockIdx.x * BN;

    v8f acc[4] = {};

    const bool fully_masked = causal && (n0 > m0 + BM - 1);
    if (!fully_masked) {
        const int ar  = tid >> 2, ac = (tid & 3) * 16;     // A: 64x64, 16/thr
        const int br2 = (tid >> 3) * 2, bc = (tid & 7) * 16; // B: 2 rows x 16/thr

        // register staging (software pipeline)
        f4  aF[4];  us8 aU[2];
        f4  bF0[4], bF1[4];  us8 bU0[2], bU1[2];

        auto loadGlobal = [&](int k0) {
            const TA* ap = A + (long)(m0 + ar) * lda + k0 + ac;
            if constexpr (std::is_same<TA, float>::value) {
                const f4* av = (const f4*)ap;
#pragma unroll
                for (int q = 0; q < 4; ++q) aF[q] = av[q];
            } else {
                const us8* av = (const us8*)ap;
#pragma unroll
                for (int q = 0; q < 2; ++q) aU[q] = av[q];
            }
            const TB* bp0 = Bm + (long)(k0 + br2) * ldb + n0 + bc;
            const TB* bp1 = bp0 + ldb;
            if constexpr (std::is_same<TB, float>::value) {
                const f4* b0 = (const f4*)bp0;
                const f4* b1 = (const f4*)bp1;
#pragma unroll
                for (int q = 0; q < 4; ++q) { bF0[q] = b0[q]; bF1[q] = b1[q]; }
            } else {
                const us8* b0 = (const us8*)bp0;
                const us8* b1 = (const us8*)bp1;
#pragma unroll
                for (int q = 0; q < 2; ++q) { bU0[q] = b0[q]; bU1[q] = b1[q]; }
            }
        };

        auto storeLds = [&]() {
            // A: pack 16 bf16 -> two b128 LDS stores
            us8 o0, o1;
            if constexpr (std::is_same<TA, float>::value) {
#pragma unroll
                for (int q = 0; q < 2; ++q)
#pragma unroll
                    for (int j = 0; j < 4; ++j) {
                        o0[q * 4 + j] = f2bf(aF[q][j]);
                        o1[q * 4 + j] = f2bf(aF[q + 2][j]);
                    }
            } else {
                o0 = aU[0]; o1 = aU[1];
            }
            *(us8*)&As[ar][ac]     = o0;
            *(us8*)&As[ar][ac + 8] = o1;
            // B: pack (k,k+1) pair -> one b32 LDS store per column
#pragma unroll
            for (int j = 0; j < 16; ++j) {
                unsigned short u0, u1;
                if constexpr (std::is_same<TB, float>::value) {
                    u0 = f2bf(bF0[j >> 2][j & 3]);
                    u1 = f2bf(bF1[j >> 2][j & 3]);
                } else {
                    u0 = bU0[j >> 3][j & 7];
                    u1 = bU1[j >> 3][j & 7];
                }
                *(unsigned*)&Bs[bc + j][br2] = (unsigned)u0 | ((unsigned)u1 << 16);
            }
        };

        loadGlobal(0);
        for (int k0 = 0; k0 < K; k0 += BK) {
            storeLds();
            __syncthreads();
            if (k0 + BK < K) loadGlobal(k0 + BK);   // overlap with WMMAs below
#pragma unroll
            for (int ks = 0; ks < BK; ks += 32) {
                // A fragment: 16x32 bf16; half-wave g: K = ks + g*16 + e (16 contiguous)
                const unsigned short* arow = &As[wm + sub][ks + g * 16];
                v16bf af = mk16(*(const us8*)arow, *(const us8*)(arow + 8));
                v16bf bf_[4];
#pragma unroll
                for (int f = 0; f < 4; ++f) {
                    // B fragment: 32x16 bf16; K = ks + g*16 + e (16 contiguous)
                    const unsigned short* brow = &Bs[wn + f * 16 + sub][ks + g * 16];
                    bf_[f] = mk16(*(const us8*)brow, *(const us8*)(brow + 8));
                }
#pragma unroll
                for (int f = 0; f < 4; ++f)
                    acc[f] = __builtin_amdgcn_wmma_f32_16x16x32_bf16(
                        false, af, false, bf_[f], (short)0, acc[f], false, false);
            }
            __syncthreads();
        }
    }

    const float NEG_INF = -__builtin_inff();
#pragma unroll
    for (int f = 0; f < 4; ++f) {
        const int nloc = n0 + wn + f * 16 + sub;
#pragma unroll
        for (int r = 0; r < 8; ++r) {
            const int mloc = m0 + wm + r + 8 * g;   // C layout: VGPR r -> M=r+8*g
            float v = acc[f][r] * alpha;
            if (causal && (nloc > mloc)) v = NEG_INF;
            C[(long)mloc * ldc + nloc] = v;
        }
    }
}

// ---------------------------------------------------------------------------
// RoPE (interleaved pairs) + L2 norm over head dim. One wave per (b,s,h) row.
// src: [B, S, heads*HD] f32 rows. dst bf16 bits:
//   transposeDst=0 -> [B, heads, S, HD]   (queries, GEMM A layout)
//   transposeDst=1 -> [B, heads, HD, S]   (keys transposed, GEMM B layout)
// ---------------------------------------------------------------------------
__global__ __launch_bounds__(256)
void rope_norm(const float* __restrict__ src, unsigned short* __restrict__ dst,
               int heads, int transposeDst)
{
    const int gw   = (blockIdx.x * 256 + threadIdx.x) >> 5;
    const int lane = threadIdx.x & 31;
    if (gw >= NB * NS * heads) return;
    const int h = gw % heads;
    const int s = (gw / heads) % NS;
    const int b = gw / (heads * NS);

    const float* row = src + (long)(b * NS + s) * (heads * HD) + (long)h * HD;
    f4 t = *(const f4*)(row + lane * 4);   // elems 4*lane.. = pairs 2*lane, 2*lane+1

    float r[4];
    float ss = 0.f;
#pragma unroll
    for (int j = 0; j < 2; ++j) {
        const int p = lane * 2 + j;                 // pair index 0..63
        const float freq = powf(10000.0f, -(float)p * (1.0f / 64.0f));
        float sn, cs;
        sincosf((float)s * freq, &sn, &cs);
        const float t0 = t[2 * j], t1 = t[2 * j + 1];
        r[2 * j]     = t0 * cs - t1 * sn;
        r[2 * j + 1] = t0 * sn + t1 * cs;
        ss += r[2 * j] * r[2 * j] + r[2 * j + 1] * r[2 * j + 1];
    }
#pragma unroll
    for (int mk = 16; mk > 0; mk >>= 1) ss += __shfl_xor(ss, mk, 32);
    const float sc = rsqrtf(ss * (1.0f / HD) + 1e-6f);

    if (transposeDst) {
        const long base = ((long)(b * heads + h) * HD + lane * 4) * NS + s;
#pragma unroll
        for (int i = 0; i < 4; ++i) dst[base + (long)i * NS] = f2bf(r[i] * sc);
    } else {
        const long base = ((long)(b * heads + h) * NS + s) * (long)HD + lane * 4;
#pragma unroll
        for (int i = 0; i < 4; ++i) dst[base + i] = f2bf(r[i] * sc);
    }
}

// ---------------------------------------------------------------------------
// In-place row softmax over S=2048 f32. One 256-thread block per row.
// ---------------------------------------------------------------------------
__global__ __launch_bounds__(256)
void softmax_rows(float* __restrict__ attn)
{
    __shared__ float red[256];
    float* p = attn + (long)blockIdx.x * NS;
    const int t = threadIdx.x;
    f4 a = *(const f4*)(p + t * 8);
    f4 b = *(const f4*)(p + t * 8 + 4);

    float m = a[0];
#pragma unroll
    for (int j = 1; j < 4; ++j) m = fmaxf(m, a[j]);
#pragma unroll
    for (int j = 0; j < 4; ++j) m = fmaxf(m, b[j]);
    red[t] = m; __syncthreads();
    for (int s = 128; s > 0; s >>= 1) {
        if (t < s) red[t] = fmaxf(red[t], red[t + s]);
        __syncthreads();
    }
    m = red[0]; __syncthreads();

    float sum = 0.f;
#pragma unroll
    for (int j = 0; j < 4; ++j) { a[j] = __expf(a[j] - m); sum += a[j]; }
#pragma unroll
    for (int j = 0; j < 4; ++j) { b[j] = __expf(b[j] - m); sum += b[j]; }
    red[t] = sum; __syncthreads();
    for (int s = 128; s > 0; s >>= 1) {
        if (t < s) red[t] += red[t + s];
        __syncthreads();
    }
    const float inv = 1.0f / red[0];
    a *= inv; b *= inv;
    *(f4*)(p + t * 8)     = a;
    *(f4*)(p + t * 8 + 4) = b;
}

// ---------------------------------------------------------------------------
extern "C" void kernel_launch(void* const* d_in, const int* in_sizes, int n_in,
                              void* d_out, int out_size, void* d_ws, size_t ws_size,
                              hipStream_t stream)
{
    (void)in_sizes; (void)n_in; (void)out_size; (void)ws_size;
    const float* x  = (const float*)d_in[0];
    const float* Wq = (const float*)d_in[1];
    const float* Wk = (const float*)d_in[2];
    const float* Wv = (const float*)d_in[3];
    const float* Wo = (const float*)d_in[4];

    float* outp = (float*)d_out;                              // [B,S,D]
    float* attn = outp + (long)NB * NS * ND;                  // [B,H,S,S]

    // workspace carve
    char* w = (char*)d_ws;
    float*          qf  = (float*)w;                              // [B*S, H*HD] f32
    w += (size_t)NB * NS * NH  * HD * sizeof(float);
    float*          kf  = (float*)w;                              // [B*S, HKV*HD] f32
    w += (size_t)NB * NS * NKV * HD * sizeof(float);
    float*          vf  = (float*)w;                              // [B*S, HKV*HD] f32
    w += (size_t)NB * NS * NKV * HD * sizeof(float);
    unsigned short* qb  = (unsigned short*)w;                     // [B,H,S,HD] bf16
    w += (size_t)NB * NS * NH  * HD * sizeof(unsigned short);
    unsigned short* kTb = (unsigned short*)w;                     // [B,HKV,HD,S] bf16
    w += (size_t)NB * NKV * HD * NS * sizeof(unsigned short);
    float*          ctx = (float*)w;                              // [B,S,H*HD] f32

    const dim3 blk(256, 1, 1);
    const float inv_sqrt_hd = 0.08838834764831845f;  // 1/sqrt(128)

    // 1) Q/K/V projections: x[4096,2048] @ W
    gemm_wmma<float, float><<<dim3((NH * HD) / 128, (NB * NS) / 64, 1), blk, 0, stream>>>(
        x, ND, 0, 0, 1, Wq, NH * HD, 0, 0, 1, qf, NH * HD, 0, 0, ND, 1, 1.0f, 0);
    gemm_wmma<float, float><<<dim3((NKV * HD) / 128, (NB * NS) / 64, 1), blk, 0, stream>>>(
        x, ND, 0, 0, 1, Wk, NKV * HD, 0, 0, 1, kf, NKV * HD, 0, 0, ND, 1, 1.0f, 0);
    gemm_wmma<float, float><<<dim3((NKV * HD) / 128, (NB * NS) / 64, 1), blk, 0, stream>>>(
        x, ND, 0, 0, 1, Wv, NKV * HD, 0, 0, 1, vf, NKV * HD, 0, 0, ND, 1, 1.0f, 0);

    // 2) RoPE + L2 norm -> bf16 (q row-major, k transposed)
    rope_norm<<<dim3((NB * NS * NH) / 8), blk, 0, stream>>>(qf, qb, NH, 0);
    rope_norm<<<dim3((NB * NS * NKV) / 8), blk, 0, stream>>>(kf, kTb, NKV, 1);

    // 3) logits = q @ kT * 1/sqrt(HD) + causal mask  (batched over z = b*H + h)
    gemm_wmma<unsigned short, unsigned short><<<dim3(NS / 128, NS / 64, NB * NH), blk, 0, stream>>>(
        qb, HD, (long)NH * NS * HD, (long)NS * HD, 1,
        kTb, NS, (long)NKV * HD * NS, (long)HD * NS, NH / NKV,
        attn, NS, (long)NH * NS * NS, (long)NS * NS,
        HD, NH, inv_sqrt_hd, 1);

    // 4) softmax in place (attn is a returned output)
    softmax_rows<<<dim3(NB * NH * NS), blk, 0, stream>>>(attn);

    // 5) ctx = attn @ v   (v read f32, GQA head mapping via bRep)
    gemm_wmma<float, float><<<dim3(1, NS / 64, NB * NH), blk, 0, stream>>>(
        attn, NS, (long)NH * NS * NS, (long)NS * NS, 1,
        vf, NKV * HD, (long)NS * NKV * HD, HD, NH / NKV,
        ctx, NH * HD, (long)NS * NH * HD, HD,
        NS, NH, 1.0f, 0);

    // 6) out = ctx @ Wo
    gemm_wmma<float, float><<<dim3(ND / 128, (NB * NS) / 64, 1), blk, 0, stream>>>(
        ctx, NH * HD, 0, 0, 1, Wo, ND, 0, 0, 1, outp, ND, 0, 0, NH * HD, 1, 1.0f, 0);
}
